// HFModule_88184268521674
// MI455X (gfx1250) — matrix operand order repaired and checked by reading, not compiled, not run on hardware
//
#include <hip/hip_runtime.h>
#include <math.h>

// ---------------------------------------------------------------------------
// Problem constants (from the reference)
// ---------------------------------------------------------------------------
#define C_DIM    1024
#define IN_DIM   1024
#define OUT_DIM  1024
#define FFT_N    513          // C/2 + 1
#define M_ROWS   8192         // 4 * 2048
#define CURV     1.0e-4f
#define EPS_REF  1.0e-7f
#define LN_EPS   1.0e-5f

typedef float  v8f  __attribute__((ext_vector_type(8)));
typedef __bf16 v8bf __attribute__((ext_vector_type(8)));
typedef __bf16 v16bf __attribute__((ext_vector_type(16)));

// ---------------------------------------------------------------------------
// Helpers
// ---------------------------------------------------------------------------
__device__ __forceinline__ float nan2num(float x) {
    if (isnan(x)) return 0.0f;
    if (isinf(x)) return x > 0.0f ? 3.4028235e38f : -3.4028235e38f;
    return x;
}

// Split fp32 into bf16 hi/lo so that (float)hi + (float)lo ~= x (16 mantissa bits)
__device__ __forceinline__ void split_bf16(float x, __bf16* hi, __bf16* lo) {
    __bf16 h = (__bf16)x;
    *hi = h;
    *lo = (__bf16)(x - (float)h);
}

union frag16 { v16bf v; v8bf h[2]; };

// ---------------------------------------------------------------------------
// Kernel 1: softmax over freq_w (513 values) -> w
// ---------------------------------------------------------------------------
__global__ __launch_bounds__(256) void softmax_kernel(const float* __restrict__ fw,
                                                      float* __restrict__ w) {
    __shared__ float red[256];
    float m = -INFINITY;
    for (int i = threadIdx.x; i < FFT_N; i += 256) m = fmaxf(m, fw[i]);
    red[threadIdx.x] = m; __syncthreads();
    for (int o = 128; o > 0; o >>= 1) {
        if (threadIdx.x < o) red[threadIdx.x] = fmaxf(red[threadIdx.x], red[threadIdx.x + o]);
        __syncthreads();
    }
    m = red[0]; __syncthreads();
    float s = 0.0f;
    for (int i = threadIdx.x; i < FFT_N; i += 256) s += expf(fw[i] - m);
    red[threadIdx.x] = s; __syncthreads();
    for (int o = 128; o > 0; o >>= 1) {
        if (threadIdx.x < o) red[threadIdx.x] += red[threadIdx.x + o];
        __syncthreads();
    }
    const float inv = 1.0f / red[0];
    for (int i = threadIdx.x; i < FFT_N; i += 256) w[i] = expf(fw[i] - m) * inv;
}

// ---------------------------------------------------------------------------
// Kernel 2: h = irfft(w), real circular-convolution filter of length C.
// h[m] = (1/C) * ( w0 + (-1)^m * w[C/2] + 2*sum_{k=1}^{C/2-1} w[k] cos(2*pi*k*m/C) )
// ---------------------------------------------------------------------------
__global__ __launch_bounds__(256) void build_h_kernel(const float* __restrict__ w,
                                                      float* __restrict__ h) {
    int m = blockIdx.x * blockDim.x + threadIdx.x;
    if (m >= C_DIM) return;
    float s = w[0] + ((m & 1) ? -w[FFT_N - 1] : w[FFT_N - 1]);
    const float twopi_over_C = 6.283185307179586f / (float)C_DIM;
    for (int k = 1; k < C_DIM / 2; ++k) {
        int ph = (k * m) & (C_DIM - 1);          // exact integer phase mod C
        s += 2.0f * w[k] * cosf(twopi_over_C * (float)ph);
    }
    h[m] = s * (1.0f / (float)C_DIM);
}

// ---------------------------------------------------------------------------
// Kernel 3: circulant weight Gt[n][k] = h[(n-k) mod C], emitted as bf16 hi/lo.
// ---------------------------------------------------------------------------
__global__ __launch_bounds__(256) void build_G_kernel(const float* __restrict__ h,
                                                      __bf16* __restrict__ Ghi,
                                                      __bf16* __restrict__ Glo) {
    int idx = blockIdx.x * blockDim.x + threadIdx.x;   // 0 .. C*C-1
    int n = idx >> 10;
    int k = idx & (C_DIM - 1);
    float v = h[(n - k) & (C_DIM - 1)];
    split_bf16(v, &Ghi[idx], &Glo[idx]);
}

// ---------------------------------------------------------------------------
// Kernel 4: generic fp32 -> bf16 hi/lo split (for x, proj_w, down_w)
// ---------------------------------------------------------------------------
__global__ __launch_bounds__(256) void split_kernel(const float* __restrict__ src,
                                                    __bf16* __restrict__ hi,
                                                    __bf16* __restrict__ lo,
                                                    int n) {
    int i = blockIdx.x * blockDim.x + threadIdx.x;
    if (i < n) split_bf16(src[i], &hi[i], &lo[i]);
}

// ---------------------------------------------------------------------------
// Kernel 5: bf16x3 WMMA GEMM.  D[M,N] = A[M,K] * W^T, operands split hi/lo.
// A*B ~= Ahi*Bhi + Ahi*Blo + Alo*Bhi  (fp32 accumulate; lo*lo term ~2^-16, dropped)
// One wave -> 32x32 output tile (2x2 accumulators) for 2x operand reuse:
// per K=32 step: 16 b128 loads feed 12 WMMAs (~24 FLOP/byte from cache).
// Block = 8 waves arranged 2(M) x 4(N) -> 64 x 128 block tile.
// Fragment layouts used here (16-bit, contiguous k = 16*(lane>>4) + elem):
//   A 16x32 : lane m = lane&15; half-wave picks K-group; elems 0..7 = K kh..kh+7,
//             elems 8..15 = K kh+8..kh+15   (two b128 loads)
//   B 32x16 : lane n = lane&15; half-wave picks K 0..15 / 16..31; B[k][n]=W[n][k]
//             -> 16 contiguous bf16 from row-major [N,K]   (two b128 loads)
//   C/D f32 : lane n = lane&15; VGPR r -> M = r + 8*(lane>>4)
// mode: 0 = silu epilogue, 1 = none, 2 = multiply by *scale_ptr
// ---------------------------------------------------------------------------
__global__ __launch_bounds__(256) void gemm_bf16x3(const __bf16* __restrict__ Ahi,
                                                   const __bf16* __restrict__ Alo,
                                                   const __bf16* __restrict__ Whi,
                                                   const __bf16* __restrict__ Wlo,
                                                   float* __restrict__ D,
                                                   int M, int N, int K,
                                                   int mode,
                                                   const float* __restrict__ scale_ptr) {
    const int lane = threadIdx.x & 31;
    const int wid  = threadIdx.x >> 5;
    const int wm   = wid >> 2;            // 0..1
    const int wn   = wid & 3;             // 0..3
    const int row0 = blockIdx.y * 64 + wm * 32;
    const int col0 = blockIdx.x * 128 + wn * 32;
    const int lm   = lane & 15;
    const int ka   = (lane >> 4) << 4;    // A K-group base: 0 or 16
    const int kb   = (lane >> 4) << 4;    // B K-group base: 0 or 16

    const __bf16* A0h = Ahi + (size_t)(row0 + lm) * K + ka;
    const __bf16* A0l = Alo + (size_t)(row0 + lm) * K + ka;
    const __bf16* A1h = A0h + (size_t)16 * K;
    const __bf16* A1l = A0l + (size_t)16 * K;
    const __bf16* W0h = Whi + (size_t)(col0 + lm) * K + kb;
    const __bf16* W0l = Wlo + (size_t)(col0 + lm) * K + kb;
    const __bf16* W1h = W0h + (size_t)16 * K;
    const __bf16* W1l = W0l + (size_t)16 * K;

    v8f acc00 = {}, acc01 = {}, acc10 = {}, acc11 = {};

    for (int k0 = 0; k0 < K; k0 += 32) {
        frag16 ah0, al0, ah1, al1, bh0, bl0, bh1, bl1;
        ah0.h[0] = *(const v8bf*)(A0h + k0);
        ah0.h[1] = *(const v8bf*)(A0h + k0 + 8);
        al0.h[0] = *(const v8bf*)(A0l + k0);
        al0.h[1] = *(const v8bf*)(A0l + k0 + 8);
        ah1.h[0] = *(const v8bf*)(A1h + k0);
        ah1.h[1] = *(const v8bf*)(A1h + k0 + 8);
        al1.h[0] = *(const v8bf*)(A1l + k0);
        al1.h[1] = *(const v8bf*)(A1l + k0 + 8);
        bh0.h[0] = *(const v8bf*)(W0h + k0);
        bh0.h[1] = *(const v8bf*)(W0h + k0 + 8);
        bl0.h[0] = *(const v8bf*)(W0l + k0);
        bl0.h[1] = *(const v8bf*)(W0l + k0 + 8);
        bh1.h[0] = *(const v8bf*)(W1h + k0);
        bh1.h[1] = *(const v8bf*)(W1h + k0 + 8);
        bl1.h[0] = *(const v8bf*)(W1l + k0);
        bl1.h[1] = *(const v8bf*)(W1l + k0 + 8);

        acc00 = __builtin_amdgcn_wmma_f32_16x16x32_bf16(false, ah0.v, false, bl0.v, (short)0, acc00, false, false);
        acc00 = __builtin_amdgcn_wmma_f32_16x16x32_bf16(false, al0.v, false, bh0.v, (short)0, acc00, false, false);
        acc00 = __builtin_amdgcn_wmma_f32_16x16x32_bf16(false, ah0.v, false, bh0.v, (short)0, acc00, false, false);

        acc01 = __builtin_amdgcn_wmma_f32_16x16x32_bf16(false, ah0.v, false, bl1.v, (short)0, acc01, false, false);
        acc01 = __builtin_amdgcn_wmma_f32_16x16x32_bf16(false, al0.v, false, bh1.v, (short)0, acc01, false, false);
        acc01 = __builtin_amdgcn_wmma_f32_16x16x32_bf16(false, ah0.v, false, bh1.v, (short)0, acc01, false, false);

        acc10 = __builtin_amdgcn_wmma_f32_16x16x32_bf16(false, ah1.v, false, bl0.v, (short)0, acc10, false, false);
        acc10 = __builtin_amdgcn_wmma_f32_16x16x32_bf16(false, al1.v, false, bh0.v, (short)0, acc10, false, false);
        acc10 = __builtin_amdgcn_wmma_f32_16x16x32_bf16(false, ah1.v, false, bh0.v, (short)0, acc10, false, false);

        acc11 = __builtin_amdgcn_wmma_f32_16x16x32_bf16(false, ah1.v, false, bl1.v, (short)0, acc11, false, false);
        acc11 = __builtin_amdgcn_wmma_f32_16x16x32_bf16(false, al1.v, false, bh1.v, (short)0, acc11, false, false);
        acc11 = __builtin_amdgcn_wmma_f32_16x16x32_bf16(false, ah1.v, false, bh1.v, (short)0, acc11, false, false);
    }

    const float scl = (mode == 2) ? scale_ptr[0] : 1.0f;
    const int mbase = (lane >> 4) << 3;   // 0 or 8
    v8f accs[2][2] = {{acc00, acc01}, {acc10, acc11}};
#pragma unroll
    for (int i = 0; i < 2; ++i) {
#pragma unroll
        for (int j = 0; j < 2; ++j) {
#pragma unroll
            for (int r = 0; r < 8; ++r) {
                float val = accs[i][j][r];
                if (mode == 0) val = val / (1.0f + expf(-val));   // silu
                else if (mode == 2) val *= scl;
                D[(size_t)(row0 + i * 16 + mbase + r) * N + (col0 + j * 16 + lm)] = val;
            }
        }
    }
}

// ---------------------------------------------------------------------------
// Kernel 6: row LayerNorm over C=1024; reads f32, emits bf16 hi/lo.
// ---------------------------------------------------------------------------
__global__ __launch_bounds__(256) void ln_kernel(const float* __restrict__ comp,
                                                 const float* __restrict__ g,
                                                 const float* __restrict__ b,
                                                 __bf16* __restrict__ ohi,
                                                 __bf16* __restrict__ olo) {
    __shared__ float rs[256], rs2[256];
    const float* p = comp + (size_t)blockIdx.x * C_DIM;
    float v[4];
    float s = 0.0f, s2 = 0.0f;
#pragma unroll
    for (int i = 0; i < 4; ++i) {
        v[i] = p[threadIdx.x + 256 * i];
        s += v[i]; s2 += v[i] * v[i];
    }
    rs[threadIdx.x] = s; rs2[threadIdx.x] = s2; __syncthreads();
    for (int o = 128; o > 0; o >>= 1) {
        if (threadIdx.x < o) {
            rs[threadIdx.x]  += rs[threadIdx.x + o];
            rs2[threadIdx.x] += rs2[threadIdx.x + o];
        }
        __syncthreads();
    }
    const float mu  = rs[0] * (1.0f / (float)C_DIM);
    const float var = rs2[0] * (1.0f / (float)C_DIM) - mu * mu;
    const float inv = rsqrtf(var + LN_EPS);
    const size_t base = (size_t)blockIdx.x * C_DIM;
#pragma unroll
    for (int i = 0; i < 4; ++i) {
        int idx = threadIdx.x + 256 * i;
        float r = (v[i] - mu) * inv * g[idx] + b[idx];
        split_bf16(r, &ohi[base + idx], &olo[base + idx]);
    }
}

// ---------------------------------------------------------------------------
// Kernel 7: Poincare ball layer; reads flat f32, emits hyp as bf16 hi/lo.
// Reductions collapse to su2, suv, sv2; Mobius ops collapse to 2 row scalars.
// ---------------------------------------------------------------------------
__global__ __launch_bounds__(256) void poincare_kernel(const float* __restrict__ flat,
                                                       const float* __restrict__ t_ptr,
                                                       __bf16* __restrict__ ohi,
                                                       __bf16* __restrict__ olo) {
    __shared__ float r0[256], r1[256], r2[256];
    __shared__ float coef[2];
    const float t = t_ptr[0];
    const float* p = flat + (size_t)blockIdx.x * C_DIM;

    float u[4], vv[4];
    float su2 = 0.0f, suv = 0.0f, sv2 = 0.0f;
#pragma unroll
    for (int i = 0; i < 4; ++i) {
        float x = nan2num(p[threadIdx.x + 256 * i]);
        float y = nan2num(t * x);
        u[i] = x; vv[i] = y;
        su2 += x * x; suv += x * y; sv2 += y * y;
    }
    r0[threadIdx.x] = su2; r1[threadIdx.x] = suv; r2[threadIdx.x] = sv2;
    __syncthreads();
    for (int o = 128; o > 0; o >>= 1) {
        if (threadIdx.x < o) {
            r0[threadIdx.x] += r0[threadIdx.x + o];
            r1[threadIdx.x] += r1[threadIdx.x + o];
            r2[threadIdx.x] += r2[threadIdx.x + o];
        }
        __syncthreads();
    }
    if (threadIdx.x == 0) {
        const float c  = CURV;
        const float sc = sqrtf(c);               // 0.01
        const float s_u2 = r0[0], s_uv = r1[0], s_v2 = r2[0];

        float nu = fmaxf(sqrtf(s_u2), EPS_REF);
        float nv = fmaxf(sqrtf(s_v2), EPS_REF);
        float scnu = fminf(fmaxf(sc * nu, EPS_REF), 1.0f - 1.0e-5f);
        float scnv = fminf(fmaxf(sc * nv, EPS_REF), 1.0f - 1.0e-5f);
        float fa = tanhf((1.0f - t) * atanhf(scnu)) / (sc * nu);
        float fb = tanhf(t * atanhf(scnv)) / (sc * nv);

        float x2 = fa * fa * s_u2;
        float y2 = fb * fb * s_v2;
        float xy = fa * fb * s_uv;
        float cA = 1.0f + 2.0f * c * xy + c * y2;
        float cB = 1.0f - c * x2;
        float den = fmaxf(1.0f + 2.0f * c * xy + c * c * x2 * y2, EPS_REF);
        coef[0] = fa * cA / den;
        coef[1] = fb * cB / den;
    }
    __syncthreads();
    const float kau = coef[0], kbv = coef[1];
    const size_t base = (size_t)blockIdx.x * C_DIM;
#pragma unroll
    for (int i = 0; i < 4; ++i) {
        int idx = threadIdx.x + 256 * i;
        float r = nan2num(kau * u[i] + kbv * vv[i]);
        split_bf16(r, &ohi[base + idx], &olo[base + idx]);
    }
}

// ---------------------------------------------------------------------------
// Launch (workspace ~76 MB):
//   [ 0,32MB)  f32 buffer: comp, then flat (GEMM2 overwrites after LN consumed it)
//   [32,48MB)  bf16 hi buffer: x_hi -> ln_hi -> hyp_hi
//   [48,64MB)  bf16 lo buffer: x_lo -> ln_lo -> hyp_lo
//   [64,68MB)  Gt hi/lo (2MB each)
//   [68,72MB)  proj_w hi/lo
//   [72,76MB)  down_w hi/lo
//   [76MB,..)  softmax(w)[513], h[1024]
// ---------------------------------------------------------------------------
extern "C" void kernel_launch(void* const* d_in, const int* in_sizes, int n_in,
                              void* d_out, int out_size, void* d_ws, size_t ws_size,
                              hipStream_t stream) {
    const float* x       = (const float*)d_in[0];
    const float* proj_w  = (const float*)d_in[1];
    const float* ln_g    = (const float*)d_in[2];
    const float* ln_b    = (const float*)d_in[3];
    const float* freq_w  = (const float*)d_in[4];
    const float* t_ptr   = (const float*)d_in[5];
    const float* down_w  = (const float*)d_in[6];
    const float* scale_p = (const float*)d_in[7];
    float*       out     = (float*)d_out;

    char* ws = (char*)d_ws;
    float*  fbuf   = (float*) (ws);
    __bf16* bhi    = (__bf16*)(ws + ((size_t)32 << 20));
    __bf16* blo    = (__bf16*)(ws + ((size_t)48 << 20));
    __bf16* Gt_hi  = (__bf16*)(ws + ((size_t)64 << 20));
    __bf16* Gt_lo  = (__bf16*)(ws + ((size_t)66 << 20));
    __bf16* pw_hi  = (__bf16*)(ws + ((size_t)68 << 20));
    __bf16* pw_lo  = (__bf16*)(ws + ((size_t)70 << 20));
    __bf16* dw_hi  = (__bf16*)(ws + ((size_t)72 << 20));
    __bf16* dw_lo  = (__bf16*)(ws + ((size_t)74 << 20));
    float*  wsm    = (float*) (ws + ((size_t)76 << 20));
    float*  hbuf   = (float*) (ws + ((size_t)76 << 20) + 4096);

    // Filter construction (tiny)
    softmax_kernel<<<1, 256, 0, stream>>>(freq_w, wsm);
    build_h_kernel<<<C_DIM / 256, 256, 0, stream>>>(wsm, hbuf);
    build_G_kernel<<<(C_DIM * C_DIM) / 256, 256, 0, stream>>>(hbuf, Gt_hi, Gt_lo);

    // Operand splits
    split_kernel<<<(M_ROWS * IN_DIM) / 256, 256, 0, stream>>>(x, bhi, blo, M_ROWS * IN_DIM);
    split_kernel<<<(C_DIM * IN_DIM) / 256, 256, 0, stream>>>(proj_w, pw_hi, pw_lo, C_DIM * IN_DIM);
    split_kernel<<<(OUT_DIM * C_DIM) / 256, 256, 0, stream>>>(down_w, dw_hi, dw_lo, OUT_DIM * C_DIM);

    dim3 gemm_block(256);
    dim3 gemm_grid(C_DIM / 128, M_ROWS / 64);   // N tiles x M tiles (8 x 128)

    // GEMM1: comp = silu(x @ proj_w^T)
    gemm_bf16x3<<<gemm_grid, gemm_block, 0, stream>>>(
        bhi, blo, pw_hi, pw_lo, fbuf, M_ROWS, C_DIM, IN_DIM, /*mode=*/0, nullptr);

    // LayerNorm -> bf16 hi/lo (reuses x's bf16 buffers)
    ln_kernel<<<M_ROWS, 256, 0, stream>>>(fbuf, ln_g, ln_b, bhi, blo);

    // GEMM2: flat = ln @ Gt^T  (== irfft(rfft(comp) * softmax(freq_w)))
    gemm_bf16x3<<<gemm_grid, gemm_block, 0, stream>>>(
        bhi, blo, Gt_hi, Gt_lo, fbuf, M_ROWS, C_DIM, C_DIM, /*mode=*/1, nullptr);

    // Poincare ball layer -> hyp as bf16 hi/lo (reuses bf16 buffers again)
    poincare_kernel<<<M_ROWS, 256, 0, stream>>>(fbuf, t_ptr, bhi, blo);

    // GEMM3: out = hyp @ down_w^T * scale
    gemm_bf16x3<<<gemm_grid, gemm_block, 0, stream>>>(
        bhi, blo, dw_hi, dw_lo, out, M_ROWS, OUT_DIM, C_DIM, /*mode=*/2, scale_p);
}
